// GroupLoss_88587995447857
// MI455X (gfx1250) — matrix operand order repaired and checked by reading, not compiled, not run on hardware
//
#include <hip/hip_runtime.h>
#include <hip/hip_bf16.h>

// ---------------------------------------------------------------------------
// GroupLoss for MI455X (gfx1250, wave32, WMMA).
// N=2048 rows, D=2048 feature dim, M=8192 classes, T=10 steps, 1024 anchors.
// ---------------------------------------------------------------------------

#define N_ROWS 2048
#define D_FEAT 2048
#define M_CLS  8192
#define T_STEPS 10
#define N_ANCH 1024
#define EPSF   1e-12f

typedef __attribute__((ext_vector_type(16))) __bf16 v16bf;
typedef __attribute__((ext_vector_type(8)))  __bf16 v8bf;
typedef __attribute__((ext_vector_type(8)))  float  v8f;

__device__ __forceinline__ __bf16 f2bf(float f) {
    unsigned u = __builtin_bit_cast(unsigned, f);
    unsigned r = u + 0x7FFFu + ((u >> 16) & 1u);      // round-to-nearest-even
    unsigned short s = (unsigned short)(r >> 16);
    return __builtin_bit_cast(__bf16, s);
}
__device__ __forceinline__ float bf2f(__bf16 b) {
    unsigned short s = __builtin_bit_cast(unsigned short, b);
    unsigned u = ((unsigned)s) << 16;
    return __builtin_bit_cast(float, u);
}

// ---------------------------------------------------------------------------
// K0: per-row mean-center + L2-normalize -> Y (bf16, N x D).
// (The features L2-normalization cancels inside Pearson; cosine of the
//  centered rows IS the correlation coefficient.)
// ---------------------------------------------------------------------------
__global__ void k_center_norm(const float* __restrict__ F, __bf16* __restrict__ Y) {
    const int row = blockIdx.x, tid = threadIdx.x;
    __shared__ float red[256];
    const float* fr = F + (size_t)row * D_FEAT;

    float s = 0.f;
    for (int j = tid; j < D_FEAT; j += 256) s += fr[j];
    red[tid] = s; __syncthreads();
    for (int o = 128; o > 0; o >>= 1) { if (tid < o) red[tid] += red[tid + o]; __syncthreads(); }
    const float mean = red[0] * (1.0f / D_FEAT);
    __syncthreads();

    float q = 0.f;
    for (int j = tid; j < D_FEAT; j += 256) { float v = fr[j] - mean; q += v * v; }
    red[tid] = q; __syncthreads();
    for (int o = 128; o > 0; o >>= 1) { if (tid < o) red[tid] += red[tid + o]; __syncthreads(); }
    const float inv = rsqrtf(fmaxf(red[0], 1e-30f));
    __syncthreads();

    __bf16* yr = Y + (size_t)row * D_FEAT;
    for (int j = tid; j < D_FEAT; j += 256) yr[j] = f2bf((fr[j] - mean) * inv);
}

// ---------------------------------------------------------------------------
// K1: W = relu(Y Y^T), zero diagonal.  bf16 WMMA 16x16x32, f32 accumulate.
// Wave computes a 32x64 tile (2 A-tiles x 4 B-tiles = 8 WMMAs per k-step).
// grid = (N/512, N/32), block = 256 (8 waves).
// Since C = Y Y^T, the B operand's column n is simply Y row n (K-contiguous).
// ---------------------------------------------------------------------------
__global__ void k_gram_relu(const __bf16* __restrict__ Y, __bf16* __restrict__ W) {
    const int tid  = threadIdx.x;
    const int wave = tid >> 5;
    const int lane = tid & 31;
    const int lo16 = lane & 15;
    const int hi   = lane >> 4;

    const int rowBase = blockIdx.y * 32;
    const int colBase = blockIdx.x * 512 + wave * 64;

    v8f acc[2][4];
    #pragma unroll
    for (int i = 0; i < 2; ++i)
        #pragma unroll
        for (int j = 0; j < 4; ++j)
            acc[i][j] = (v8f){0.f,0.f,0.f,0.f,0.f,0.f,0.f,0.f};

    for (int k = 0; k < D_FEAT; k += 32) {
        v16bf a[2];
        #pragma unroll
        for (int i = 0; i < 2; ++i) {
            // A tile (16x32): lanes 0-15 hold K {0..15}; lanes 16-31 K {16..31}
            const __bf16* ap = Y + (size_t)(rowBase + i * 16 + lo16) * D_FEAT + k + hi * 16;
            v8bf alo = *(const v8bf*)(ap);
            v8bf ahi = *(const v8bf*)(ap + 8);
            a[i] = __builtin_shufflevector(alo, ahi, 0,1,2,3,4,5,6,7,8,9,10,11,12,13,14,15);
        }
        #pragma unroll
        for (int j = 0; j < 4; ++j) {
            // B tile (32x16): lane = column N, (lane>>4) selects K half; K-contiguous 16
            const __bf16* bp = Y + (size_t)(colBase + j * 16 + lo16) * D_FEAT + k + hi * 16;
            v16bf b = *(const v16bf*)(bp);
            acc[0][j] = __builtin_amdgcn_wmma_f32_16x16x32_bf16(
                           false, a[0], false, b, (short)0, acc[0][j], false, false);
            acc[1][j] = __builtin_amdgcn_wmma_f32_16x16x32_bf16(
                           false, a[1], false, b, (short)0, acc[1][j], false, false);
        }
    }

    #pragma unroll
    for (int i = 0; i < 2; ++i)
        #pragma unroll
        for (int j = 0; j < 4; ++j)
            #pragma unroll
            for (int r = 0; r < 8; ++r) {
                const int gR = rowBase + i * 16 + r + hi * 8;
                const int gC = colBase + j * 16 + lo16;
                float v = acc[i][j][r];
                v = (gR == gC) ? 0.f : fmaxf(v, 0.f);
                W[(size_t)gR * N_ROWS + gC] = f2bf(v);
            }
}

// ---------------------------------------------------------------------------
// K2: isolated-row fix: if sum(W[row,:]) == 0 -> W[row,row] = 1
// ---------------------------------------------------------------------------
__global__ void k_rowfix(__bf16* __restrict__ W) {
    const int row = blockIdx.x, tid = threadIdx.x;
    __shared__ float red[256];
    float s = 0.f;
    const __bf16* wr = W + (size_t)row * N_ROWS;
    for (int j = tid; j < N_ROWS; j += 256) s += bf2f(wr[j]);
    red[tid] = s; __syncthreads();
    for (int o = 128; o > 0; o >>= 1) { if (tid < o) red[tid] += red[tid + o]; __syncthreads(); }
    if (tid == 0 && red[0] <= 0.f) W[(size_t)row * N_ROWS + row] = f2bf(1.0f);
}

// ---------------------------------------------------------------------------
// K3: row softmax of X -> Xc (f32, N x M)
// ---------------------------------------------------------------------------
__global__ void k_softmax(const float* __restrict__ X, float* __restrict__ Xc) {
    const int row = blockIdx.x, tid = threadIdx.x;
    __shared__ float red[256];
    const float* xr = X + (size_t)row * M_CLS;
    float m = -3.4e38f;
    for (int j = tid; j < M_CLS; j += 256) m = fmaxf(m, xr[j]);
    red[tid] = m; __syncthreads();
    for (int o = 128; o > 0; o >>= 1) { if (tid < o) red[tid] = fmaxf(red[tid], red[tid + o]); __syncthreads(); }
    m = red[0]; __syncthreads();

    float s = 0.f;
    for (int j = tid; j < M_CLS; j += 256) s += expf(xr[j] - m);
    red[tid] = s; __syncthreads();
    for (int o = 128; o > 0; o >>= 1) { if (tid < o) red[tid] += red[tid + o]; __syncthreads(); }
    const float inv = 1.0f / red[0];

    float* cr = Xc + (size_t)row * M_CLS;
    for (int j = tid; j < M_CLS; j += 256) cr[j] = expf(xr[j] - m) * inv;
}

// ---------------------------------------------------------------------------
// Anchor flags / rowsum init
// ---------------------------------------------------------------------------
__global__ void k_flags_clear(int* __restrict__ flags) {
    int i = blockIdx.x * blockDim.x + threadIdx.x;
    if (i < N_ROWS) flags[i] = 0;
}
__global__ void k_flags_set(const int* __restrict__ anchors, int* __restrict__ flags) {
    int i = blockIdx.x * blockDim.x + threadIdx.x;
    if (i < N_ANCH) flags[anchors[i]] = 1;
}
__global__ void k_ones_rowsum(float* __restrict__ rowsum) {
    int i = blockIdx.x * blockDim.x + threadIdx.x;
    if (i < N_ROWS) rowsum[i] = 1.0f;
}
__global__ void k_zero_rowsum(float* __restrict__ rowsum) {
    int i = blockIdx.x * blockDim.x + threadIdx.x;
    if (i < N_ROWS) rowsum[i] = 0.f;
}

// ---------------------------------------------------------------------------
// K4 (fused): normalize Xc by previous row-sums, apply anchor clamp, and write
// XbT (bf16, M x N) through an LDS 64x64 tile so both the f32 reads and the
// bf16 transposed writes are coalesced.  grid = (M/64, N/64), block = 256.
// ---------------------------------------------------------------------------
__global__ void k_norm_build_xbt(float* __restrict__ Xc, const float* __restrict__ rowsum,
                                 const int* __restrict__ flags, const int* __restrict__ targets,
                                 __bf16* __restrict__ XbT) {
    __shared__ __bf16 tile[64][72];   // padded to dodge bank conflicts
    const int tid = threadIdx.x;
    const int colBase = blockIdx.x * 64;
    const int rowBase = blockIdx.y * 64;

    // load + normalize + clamp: 64 threads span a 64-col row segment (coalesced)
    #pragma unroll
    for (int it = 0; it < 16; ++it) {
        const int li = it * 256 + tid;
        const int r = li >> 6;          // 0..63
        const int c = li & 63;          // 0..63
        const int row = rowBase + r;
        const int col = colBase + c;
        const float inv = 1.0f / fmaxf(rowsum[row], EPSF);
        const size_t idx = (size_t)row * M_CLS + col;
        const float v = Xc[idx] * inv;
        Xc[idx] = v;
        __bf16 b;
        if (flags[row]) b = f2bf((col == targets[row]) ? 1.0f : 0.0f);
        else            b = f2bf(v);
        tile[r][c] = b;
    }
    __syncthreads();

    // transposed store: 64 threads span 64 consecutive rows of Xb (= contiguous
    // elements of an XbT row) -> 128B coalesced bf16 writes
    #pragma unroll
    for (int it = 0; it < 16; ++it) {
        const int li = it * 256 + tid;
        const int cr = li >> 6;         // col-in-tile
        const int rr = li & 63;         // row-in-tile
        XbT[(size_t)(colBase + cr) * N_ROWS + rowBase + rr] = tile[rr][cr];
    }
}

// ---------------------------------------------------------------------------
// K5: fused step GEMM: Pi = W @ Xb (bf16 WMMA, f32 acc), PX = Xc * Pi written
// back to Xc in place, rowsum += sum_j |PX|.
// Wave computes a 32x64 tile (8 WMMAs / k-step).  grid = (M/512, N/32).
// ---------------------------------------------------------------------------
__global__ void k_step_gemm(const __bf16* __restrict__ W, const __bf16* __restrict__ XbT,
                            float* __restrict__ Xc, float* __restrict__ rowsum) {
    const int tid  = threadIdx.x;
    const int wave = tid >> 5;
    const int lane = tid & 31;
    const int lo16 = lane & 15;
    const int hi   = lane >> 4;

    const int rowBase = blockIdx.y * 32;
    const int colBase = blockIdx.x * 512 + wave * 64;

    __shared__ float rs[8][32];
    rs[wave][lane] = 0.f;
    __syncthreads();

    v8f acc[2][4];
    #pragma unroll
    for (int i = 0; i < 2; ++i)
        #pragma unroll
        for (int j = 0; j < 4; ++j)
            acc[i][j] = (v8f){0.f,0.f,0.f,0.f,0.f,0.f,0.f,0.f};

    for (int k = 0; k < N_ROWS; k += 32) {
        v16bf a[2];
        #pragma unroll
        for (int i = 0; i < 2; ++i) {
            const __bf16* ap = W + (size_t)(rowBase + i * 16 + lo16) * N_ROWS + k + hi * 16;
            v8bf alo = *(const v8bf*)(ap);
            v8bf ahi = *(const v8bf*)(ap + 8);
            a[i] = __builtin_shufflevector(alo, ahi, 0,1,2,3,4,5,6,7,8,9,10,11,12,13,14,15);
        }
        #pragma unroll
        for (int j = 0; j < 4; ++j) {
            const __bf16* bp = XbT + (size_t)(colBase + j * 16 + lo16) * N_ROWS + k + hi * 16;
            v16bf b = *(const v16bf*)(bp);
            acc[0][j] = __builtin_amdgcn_wmma_f32_16x16x32_bf16(
                           false, a[0], false, b, (short)0, acc[0][j], false, false);
            acc[1][j] = __builtin_amdgcn_wmma_f32_16x16x32_bf16(
                           false, a[1], false, b, (short)0, acc[1][j], false, false);
        }
    }

    // fused epilogue: PX = Xc * Pi (in place) + per-row |PX| partial sums
    float racc[2][8];
    #pragma unroll
    for (int i = 0; i < 2; ++i)
        #pragma unroll
        for (int r = 0; r < 8; ++r) racc[i][r] = 0.f;

    #pragma unroll
    for (int i = 0; i < 2; ++i)
        #pragma unroll
        for (int j = 0; j < 4; ++j) {
            const int gC = colBase + j * 16 + lo16;
            #pragma unroll
            for (int r = 0; r < 8; ++r) {
                const int gR = rowBase + i * 16 + r + hi * 8;
                const size_t idx = (size_t)gR * M_CLS + gC;
                float px = Xc[idx] * acc[i][j][r];
                Xc[idx] = px;
                racc[i][r] += fabsf(px);
            }
        }

    #pragma unroll
    for (int i = 0; i < 2; ++i)
        #pragma unroll
        for (int r = 0; r < 8; ++r)
            atomicAdd(&rs[wave][i * 16 + r + hi * 8], racc[i][r]);
    __syncthreads();

    if (tid < 32) {
        float t = 0.f;
        #pragma unroll
        for (int w = 0; w < 8; ++w) t += rs[w][tid];
        atomicAdd(&rowsum[rowBase + tid], t);
    }
}

// ---------------------------------------------------------------------------
// K7: loss = -mean(log(clip(Xc[row,targets[row]] / max(rowsum[row],EPS), EPS)))
// (final step's normalization folded in — only the gathered entries matter)
// ---------------------------------------------------------------------------
__global__ void k_loss(const float* __restrict__ Xc, const float* __restrict__ rowsum,
                       const int* __restrict__ targets, float* __restrict__ out) {
    const int tid = threadIdx.x;
    __shared__ float red[256];
    float s = 0.f;
    for (int row = tid; row < N_ROWS; row += 256) {
        float v = Xc[(size_t)row * M_CLS + targets[row]] / fmaxf(rowsum[row], EPSF);
        s += logf(fmaxf(v, EPSF));
    }
    red[tid] = s; __syncthreads();
    for (int o = 128; o > 0; o >>= 1) { if (tid < o) red[tid] += red[tid + o]; __syncthreads(); }
    if (tid == 0) out[0] = -red[0] * (1.0f / N_ROWS);
}

// ---------------------------------------------------------------------------
// Host-side orchestration
// ---------------------------------------------------------------------------
extern "C" void kernel_launch(void* const* d_in, const int* in_sizes, int n_in,
                              void* d_out, int out_size, void* d_ws, size_t ws_size,
                              hipStream_t stream) {
    (void)in_sizes; (void)n_in; (void)out_size; (void)ws_size;

    const float* features = (const float*)d_in[0];   // (2048, 2048) f32
    const float* X        = (const float*)d_in[1];   // (2048, 8192) f32
    const int*   targets  = (const int*)d_in[2];     // (2048,) i32
    const int*   anchors  = (const int*)d_in[3];     // (1024,) i32
    float*       out      = (float*)d_out;           // scalar

    // workspace carve-up
    char* ws = (char*)d_ws;
    size_t off = 0;
    __bf16* Y      = (__bf16*)(ws + off); off += (size_t)N_ROWS * D_FEAT * 2;  //  8 MB
    __bf16* W      = (__bf16*)(ws + off); off += (size_t)N_ROWS * N_ROWS * 2;  //  8 MB
    float*  Xc     = (float*) (ws + off); off += (size_t)N_ROWS * M_CLS  * 4;  // 64 MB
    __bf16* XbT    = (__bf16*)(ws + off); off += (size_t)M_CLS  * N_ROWS * 2;  // 32 MB
    float*  rowsum = (float*) (ws + off); off += (size_t)N_ROWS * 4;
    int*    flags  = (int*)   (ws + off); off += (size_t)N_ROWS * 4;

    // --- build transition matrix W ---
    k_center_norm<<<N_ROWS, 256, 0, stream>>>(features, Y);
    k_gram_relu<<<dim3(N_ROWS / 512, N_ROWS / 32), 256, 0, stream>>>(Y, W);
    k_rowfix<<<N_ROWS, 256, 0, stream>>>(W);

    // --- anchor flags + initial softmax ---
    k_flags_clear<<<N_ROWS / 256, 256, 0, stream>>>(flags);
    k_flags_set<<<N_ANCH / 256, 256, 0, stream>>>(anchors, flags);
    k_softmax<<<N_ROWS, 256, 0, stream>>>(X, Xc);
    k_ones_rowsum<<<N_ROWS / 256, 256, 0, stream>>>(rowsum);

    // --- T replicator steps (normalization of step t-1 fused into build of t) ---
    for (int t = 0; t < T_STEPS; ++t) {
        k_norm_build_xbt<<<dim3(M_CLS / 64, N_ROWS / 64), 256, 0, stream>>>(
            Xc, rowsum, flags, targets, XbT);
        k_zero_rowsum<<<N_ROWS / 256, 256, 0, stream>>>(rowsum);
        k_step_gemm<<<dim3(M_CLS / 512, N_ROWS / 32), 256, 0, stream>>>(W, XbT, Xc, rowsum);
    }

    // --- final NLL (folds in last step's normalization) ---
    k_loss<<<1, 256, 0, stream>>>(Xc, rowsum, targets, out);
}
